// EAGLE_V3_64089501991529
// MI455X (gfx1250) — compile-verified, matrix operand order checked by reading, not run on hardware
//
#include <hip/hip_runtime.h>

// ---------------------------------------------------------------------------
// GAT layer on gfx1250: Wh = h@W + pos ; attn = softmax(mask(leaky(f_s+f_d)))
//                       h' = attn @ Wh
// bf16 WMMA (V_WMMA_F32_16X16X32_BF16) for both GEMMs, f32 accumulate.
// Double-buffered LDS A-staging in GEMM1 (1 barrier / K-step).
// ---------------------------------------------------------------------------

typedef __bf16 bf16_t;
typedef bf16_t v16bf __attribute__((ext_vector_type(16)));
typedef bf16_t v8bf  __attribute__((ext_vector_type(8)));
typedef bf16_t v4bf  __attribute__((ext_vector_type(4)));
typedef float  v8f   __attribute__((ext_vector_type(8)));
typedef float  v4f   __attribute__((ext_vector_type(4)));

#define GAT_B     256
#define GAT_N     256
#define GAT_DIN   768
#define GAT_DO    300
#define GAT_DOP   320          // padded to 20 WMMA tiles of 16
#define GAT_KS1   (GAT_DIN / 32)
#define NEG_INF   (-9e15f)

static __device__ __forceinline__ v16bf mk16(v8bf lo, v8bf hi) {
    v16bf r;
#pragma unroll
    for (int i = 0; i < 8; ++i) { r[i] = lo[i]; r[i + 8] = hi[i]; }
    return r;
}

static __device__ __forceinline__ v8f wmma_bf16(v16bf a, v16bf b, v8f c) {
    return __builtin_amdgcn_wmma_f32_16x16x32_bf16(
        /*neg_a=*/false, a, /*neg_b=*/false, b,
        /*c_mod=*/(short)0, c, /*reuse_a=*/false, /*reuse_b=*/false);
}

// ---------------------------------------------------------------------------
// Kernel 0: W [768,300] f32 -> Wt [320,768] bf16 (transposed, zero padded)
// ---------------------------------------------------------------------------
__global__ void gat_wconv(const float* __restrict__ W, bf16_t* __restrict__ Wt) {
    int idx = blockIdx.x * 256 + threadIdx.x;      // over 768*320
    if (idx >= GAT_DIN * GAT_DOP) return;
    int k = idx / GAT_DOP;
    int n = idx - k * GAT_DOP;
    float v = (n < GAT_DO) ? W[k * GAT_DO + n] : 0.0f;
    Wt[n * GAT_DIN + k] = (bf16_t)v;
}

// ---------------------------------------------------------------------------
// Kernel 1: Wh = h@W + pos_table[positions] ; store transposed bf16
//   grid  = (B*N)/16 blocks, block = 128 threads (4 waves)
//   each wave owns 5 n-tiles of 16 (20 tiles = 320 padded cols)
//   Whb layout: [b][n (0..319)][j (0..255)] bf16
//   A tile staged f32->bf16 through a ping-pong LDS buffer so the global
//   load for step k+1 overlaps the WMMAs of step k; one barrier per step.
// ---------------------------------------------------------------------------
__global__ __launch_bounds__(128) void gat_gemm1(
    const float* __restrict__ h,          // [B*N, 768]
    const int*   __restrict__ positions,  // [B*N]
    const float* __restrict__ pos_table,  // [256, 300]
    const bf16_t* __restrict__ Wt,        // [320, 768]
    bf16_t* __restrict__ Whb)             // [B, 320, 256]
{
    __shared__ float  ldsPos[16][GAT_DOP];    // 20 KB
    __shared__ bf16_t ldsA[2][16 * 32];       // 2 KB ping-pong A tile

    const int tid  = threadIdx.x;
    const int lane = tid & 31;
    const int wave = tid >> 5;
    const int r0   = blockIdx.x * 16;         // global row base (b*256 + j0)
    const int b    = r0 >> 8;
    const int j0   = r0 & 255;

    // stage positional embeddings for the 16 rows (zero pad n>=300)
    for (int m = 0; m < 16; ++m) {
        int p = positions[r0 + m];
        for (int n = tid; n < GAT_DOP; n += 128)
            ldsPos[m][n] = (n < GAT_DO) ? pos_table[p * GAT_DO + n] : 0.0f;
    }

    v8f acc[5];
#pragma unroll
    for (int t = 0; t < 5; ++t)
#pragma unroll
        for (int k = 0; k < 8; ++k) acc[t][k] = 0.0f;

    const int row16 = lane & 15;             // A row / B column within tile
    const int khalf = (lane >> 4) << 3;      // 0 or 8
    const int mbase = (lane >> 4) << 3;      // C rows k+0..7 or k+8..15

    // A staging: thread loads 4 consecutive f32 of the 16x32 tile
    const int am = tid >> 3;                 // row 0..15
    const int ak = (tid & 7) << 2;           // col 0,4,...,28
    const float* hrow = h + (size_t)(r0 + am) * GAT_DIN + ak;

    // prologue: stage K-step 0 into buffer 0
    {
        v4f av = *(const v4f*)(hrow);
        v4bf ab;
#pragma unroll
        for (int q = 0; q < 4; ++q) ab[q] = (bf16_t)av[q];
        *(v4bf*)&ldsA[0][am * 32 + ak] = ab;
    }
    __syncthreads();

    for (int ks = 0; ks < GAT_KS1; ++ks) {
        const int kk  = ks * 32;
        const int cur = ks & 1;
        // issue the load for the NEXT K-step first (clamped, branch-free)
        const int kn = (ks + 1 < GAT_KS1) ? (ks + 1) : ks;
        v4f avn = *(const v4f*)(hrow + kn * 32);

        // A fragment for this step: two 16B chunks per lane
        v8bf alo = *(const v8bf*)&ldsA[cur][row16 * 32 + khalf];
        v8bf ahi = *(const v8bf*)&ldsA[cur][row16 * 32 + khalf + 16];
        v16bf afrag = mk16(alo, ahi);

#pragma unroll
        for (int t = 0; t < 5; ++t) {
            const int n = (wave * 5 + t) * 16 + row16;
            const bf16_t* bp = Wt + (size_t)n * GAT_DIN + kk + khalf;
            v8bf blo = *(const v8bf*)bp;
            v8bf bhi = *(const v8bf*)(bp + 16);
            acc[t] = wmma_bf16(afrag, mk16(blo, bhi), acc[t]);
        }

        // stage next tile into the other buffer (safe: it was last read in
        // step ks-1, which is fenced by the previous barrier)
        v4bf abn;
#pragma unroll
        for (int q = 0; q < 4; ++q) abn[q] = (bf16_t)avn[q];
        *(v4bf*)&ldsA[cur ^ 1][am * 32 + ak] = abn;
        __syncthreads();
    }

    // epilogue: add pos embedding, convert to bf16, store transposed:
    // lane holds column n fixed, rows j0+mbase .. j0+mbase+7 -> one 16B store
#pragma unroll
    for (int t = 0; t < 5; ++t) {
        const int n = (wave * 5 + t) * 16 + row16;
        v8bf ob;
#pragma unroll
        for (int k = 0; k < 8; ++k) {
            float v = acc[t][k] + ldsPos[mbase + k][n];
            ob[k] = (bf16_t)v;
        }
        *(v8bf*)(Whb + ((size_t)(b * GAT_DOP + n) * GAT_N) + j0 + mbase) = ob;
    }
}

// ---------------------------------------------------------------------------
// Kernel 2: f_src[b,j] = sum_n Wh[b,j,n]*a_src[n]  (and f_dst)
//   Whb is [b][n][j] so column reads are fully coalesced across threads j.
// ---------------------------------------------------------------------------
__global__ __launch_bounds__(256) void gat_fvec(
    const bf16_t* __restrict__ Whb,
    const float* __restrict__ a_src, const float* __restrict__ a_dst,
    float* __restrict__ f_src, float* __restrict__ f_dst)
{
    __shared__ float as[GAT_DO], ad[GAT_DO];
    const int b = blockIdx.x, j = threadIdx.x;
    for (int n = threadIdx.x; n < GAT_DO; n += 256) {
        as[n] = a_src[n];
        ad[n] = a_dst[n];
    }
    __syncthreads();
    const bf16_t* base = Whb + (size_t)b * GAT_DOP * GAT_N + j;
    float s = 0.0f, d = 0.0f;
    for (int n = 0; n < GAT_DO; ++n) {
        float w = (float)base[(size_t)n * GAT_N];
        s += w * as[n];
        d += w * ad[n];
    }
    f_src[b * GAT_N + j] = s;
    f_dst[b * GAT_N + j] = d;
}

// ---------------------------------------------------------------------------
// Kernel 3: fused masked softmax + h' = attn @ Wh
//   grid = B * (N/16), block = 128 (4 waves).
//   Phase 1: 8 lanes per row compute masked leaky scores + softmax
//            (shfl_xor reductions), write attention f32 to d_out and
//            bf16 copy to LDS as the A matrix.
//   Phase 2: 8 K-steps of bf16 WMMA against Whb (resident in L2).
// ---------------------------------------------------------------------------
__global__ __launch_bounds__(128) void gat_attn(
    const float* __restrict__ adj,        // [B,256,256]
    const float* __restrict__ f_src,
    const float* __restrict__ f_dst,
    const bf16_t* __restrict__ Whb,       // [B,320,256]
    float* __restrict__ hprime,           // [B,256,300]
    float* __restrict__ attn_out)         // [B,256,256]
{
    __shared__ bf16_t attb[16][GAT_N];    // 8 KB, A matrix for GEMM2
    __shared__ float  fdLds[GAT_N];
    __shared__ float  fsLds[16];

    const int tid  = threadIdx.x;
    const int lane = tid & 31;
    const int wave = tid >> 5;
    const int b    = blockIdx.x >> 4;
    const int i0   = (blockIdx.x & 15) * 16;

    for (int j = tid; j < GAT_N; j += 128) fdLds[j] = f_dst[b * GAT_N + j];
    if (tid < 16) fsLds[tid] = f_src[b * GAT_N + i0 + tid];
    __syncthreads();

    // ---- phase 1: softmax over j for 16 rows, 8 lanes per row ----
    const int row = wave * 4 + (lane >> 3);     // 0..15
    const int sub = lane & 7;
    const int gi  = i0 + row;
    const float* adjrow = adj + ((size_t)(b * GAT_N + gi)) * GAT_N;
    const float fs = fsLds[row];

    float vals[32];
    float mx = -__builtin_inff();
#pragma unroll
    for (int t = 0; t < 32; ++t) {
        int j = sub + t * 8;
        float e = fs + fdLds[j];
        e = (e >= 0.0f) ? e : 0.2f * e;          // LeakyReLU(0.2)
        float m = (adjrow[j] > 0.0f) ? e : NEG_INF;
        vals[t] = m;
        mx = fmaxf(mx, m);
    }
#pragma unroll
    for (int o = 1; o < 8; o <<= 1) mx = fmaxf(mx, __shfl_xor(mx, o, 32));

    float sum = 0.0f;
#pragma unroll
    for (int t = 0; t < 32; ++t) {
        float ex = __expf(vals[t] - mx);
        vals[t] = ex;
        sum += ex;
    }
#pragma unroll
    for (int o = 1; o < 8; o <<= 1) sum += __shfl_xor(sum, o, 32);
    const float inv = 1.0f / sum;                // self-loop guarantees sum >= 1

    float* arow = attn_out + ((size_t)(b * GAT_N + gi)) * GAT_N;
#pragma unroll
    for (int t = 0; t < 32; ++t) {
        int j = sub + t * 8;
        float a = vals[t] * inv;
        arow[j] = a;
        attb[row][j] = (bf16_t)a;
    }
    __syncthreads();

    // ---- phase 2: h'[16 x 300] = attb[16 x 256] @ Wh[256 x 300] ----
    v8f acc[5];
#pragma unroll
    for (int t = 0; t < 5; ++t)
#pragma unroll
        for (int k = 0; k < 8; ++k) acc[t][k] = 0.0f;

    const int row16 = lane & 15;
    const int khalf = (lane >> 4) << 3;
    const int mbase = (lane >> 4) << 3;

#pragma unroll
    for (int ks = 0; ks < GAT_N / 32; ++ks) {
        const int kk = ks * 32;
        v8bf alo = *(const v8bf*)&attb[row16][kk + khalf];
        v8bf ahi = *(const v8bf*)&attb[row16][kk + khalf + 16];
        v16bf afrag = mk16(alo, ahi);
#pragma unroll
        for (int t = 0; t < 5; ++t) {
            const int n = (wave * 5 + t) * 16 + row16;
            const bf16_t* bp = Whb + ((size_t)(b * GAT_DOP + n)) * GAT_N + kk + khalf;
            v8bf blo = *(const v8bf*)bp;
            v8bf bhi = *(const v8bf*)(bp + 16);
            acc[t] = wmma_bf16(afrag, mk16(blo, bhi), acc[t]);
        }
    }

    // store h' (f32); lanes 0-15 / 16-31 each write 64B-contiguous per k
#pragma unroll
    for (int t = 0; t < 5; ++t) {
        const int n = (wave * 5 + t) * 16 + row16;
        if (n < GAT_DO) {
#pragma unroll
            for (int k = 0; k < 8; ++k) {
                hprime[((size_t)(b * GAT_N) + i0 + mbase + k) * GAT_DO + n] = acc[t][k];
            }
        }
    }
}

// ---------------------------------------------------------------------------
// Launch
//   inputs: 0=h 1=adj 2=positions 3=W 4=a_src 5=a_dst 6=pos_table
//   d_out : h_prime [B,N,300] then attention [B,N,N]
// ---------------------------------------------------------------------------
extern "C" void kernel_launch(void* const* d_in, const int* in_sizes, int n_in,
                              void* d_out, int out_size, void* d_ws, size_t ws_size,
                              hipStream_t stream) {
    const float* h         = (const float*)d_in[0];
    const float* adj       = (const float*)d_in[1];
    const int*   positions = (const int*)d_in[2];
    const float* W         = (const float*)d_in[3];
    const float* a_src     = (const float*)d_in[4];
    const float* a_dst     = (const float*)d_in[5];
    const float* pos_table = (const float*)d_in[6];

    float* hprime   = (float*)d_out;
    float* attn_out = hprime + (size_t)GAT_B * GAT_N * GAT_DO;

    // workspace layout
    char* ws = (char*)d_ws;
    bf16_t* Wt   = (bf16_t*)ws;                                   // 320*768*2   = 480 KB
    ws += (size_t)GAT_DOP * GAT_DIN * sizeof(bf16_t);
    bf16_t* Whb  = (bf16_t*)ws;                                   // 256*320*256*2 = 40 MB
    ws += (size_t)GAT_B * GAT_DOP * GAT_N * sizeof(bf16_t);
    float*  fsrc = (float*)ws;  ws += (size_t)GAT_B * GAT_N * sizeof(float);
    float*  fdst = (float*)ws;

    gat_wconv<<<(GAT_DIN * GAT_DOP + 255) / 256, 256, 0, stream>>>(W, Wt);
    gat_gemm1<<<(GAT_B * GAT_N) / 16, 128, 0, stream>>>(h, positions, pos_table, Wt, Whb);
    gat_fvec<<<GAT_B, 256, 0, stream>>>(Whb, a_src, a_dst, fsrc, fdst);
    gat_attn<<<GAT_B * (GAT_N / 16), 128, 0, stream>>>(adj, fsrc, fdst, Whb, hprime, attn_out);
}
